// PointTransformerBackbone_63840393888274
// MI455X (gfx1250) — matrix operand order repaired and checked by reading, not compiled, not run on hardware
//
#include <hip/hip_runtime.h>

// ============================================================================
// Point Transformer backbone for MI455X (gfx1250, wave32, WMMA).
//
// Roofline: dominant cost is the per-(point,neighbor) MLPs (delta2/gamma1/
// gamma2: 768x768 GEMMs over B*N*K = 81920 rows at stage 0) => ~0.3 TFLOP of
// dense GEMM vs ~1 GB of traffic => compute bound. All linear layers run on
// the WMMA pipe (v_wmma_f32_16x16x32_f16, fp32 accumulate). Weights are
// pre-packed to zero-padded f16 pairs so the WMMA inner loop has NO bounds
// checks and loads B as raw dwords; A is loaded as 128-bit vectors on the
// aligned (K%32==0) path and one A fragment is reused across a 16x64 output
// tile (4 WMMAs per K-step) when Nd%64==0. Irregular work (FPS, kNN, gathers,
// softmax, BN, pooling) is bandwidth trivial and runs as VALU kernels.
// ============================================================================

typedef __attribute__((ext_vector_type(16))) _Float16 v16h;
typedef __attribute__((ext_vector_type(8)))  float    v8f;
typedef __attribute__((ext_vector_type(8)))  unsigned v8u;

union BFrag { v8u u; v16h h; };

#define KMAX 20

static inline int cdiv_i(int a, int b) { return (a + b - 1) / b; }

// ---------------------------------------------------------------------------
// Weight pre-pack: W (Kd x Nd fp32) -> Wp ((Kpad/2) x Nd) of packed f16 pairs
// Wp[p*Nd + col] = { f16(W[2p,col]), f16(W[2p+1,col]) }, zero beyond Kd.
// ---------------------------------------------------------------------------
__global__ void pack_w_kernel(const float* __restrict__ W, unsigned* __restrict__ Wp,
                              int Kd, int Nd, int Kpad)
{
  int t = blockIdx.x * blockDim.x + threadIdx.x;
  int total = (Kpad >> 1) * Nd;
  if (t >= total) return;
  int col = t % Nd, p = t / Nd;
  int k0 = 2 * p, k1 = 2 * p + 1;
  union { _Float16 h[2]; unsigned u; } cv;
  cv.h[0] = (k0 < Kd) ? (_Float16)W[(size_t)k0 * Nd + col] : (_Float16)0.f;
  cv.h[1] = (k1 < Kd) ? (_Float16)W[(size_t)k1 * Nd + col] : (_Float16)0.f;
  Wp[t] = cv.u;
}

// ---------------------------------------------------------------------------
// WMMA GEMM: out[M,Nd] = act(A[M,Kd] @ W[Kd,Nd] + bias (+ resid))
// One wave per 16 x (16*NT) output tile; M%16==0, Nd%(16*NT)==0.
// B comes from the packed/padded weight buffer => branch-free.
// MODE 0: A from memory (row-major). KAL => Kd%32==0, vector loads, no guards.
// MODE 1: A row (b,n,j) synthesized: q[b,n,:] - k[b,idx,:] + pos[b,n,j,:],
//         Kd=768 (always aligned).
// A-fragment element map (ISA 7.12.2): av[j]   <- K = kb + (lane>>4)*8 + j
//                                      av[8+j] <- K = kb + 16 + (lane>>4)*8 + j
// ---------------------------------------------------------------------------
template <int MODE, bool KAL, int NT>
__global__ __launch_bounds__(32)
void gemm_wmma_t(const float* __restrict__ A, const unsigned* __restrict__ Wp,
                 const float* __restrict__ bias, const float* __restrict__ resid,
                 float* __restrict__ out,
                 int M, int Kd, int Nd, int do_relu,
                 const float* __restrict__ qv, const float* __restrict__ kv,
                 const float* __restrict__ pv, const int* __restrict__ nidx,
                 int Npts, int Kn)
{
  const int lane  = threadIdx.x;
  const int mrow  = lane & 15;
  const int hl    = lane >> 4;
  const int tileM = blockIdx.x * 16;
  const int tileN = blockIdx.y * (16 * NT);
  const int row   = tileM + mrow;
  const int col0  = tileN + mrow;
  const int Kpad  = (Kd + 31) & ~31;

  const float* arow = nullptr;
  const float* qrow = nullptr; const float* krow = nullptr; const float* prow = nullptr;
  if (MODE == 0) {
    arow = A + (size_t)row * Kd;
  } else {
    int j  = row % Kn;
    int n  = (row / Kn) % Npts;
    int b  = row / (Kn * Npts);
    int nb = nidx[((size_t)b * Npts + n) * Kn + j];
    qrow = qv + ((size_t)b * Npts + n) * 768;
    krow = kv + ((size_t)b * Npts + nb) * 768;
    prow = pv + (size_t)row * 768;
  }

  v8f acc[NT];
#pragma unroll
  for (int nt = 0; nt < NT; ++nt) acc[nt] = {};

  for (int kb = 0; kb < Kpad; kb += 32) {
    // ---- A fragment (shared across all NT column subtiles)
    v16h av;
    if (MODE == 1) {
      const int o0 = kb + hl * 8, o1 = kb + 16 + hl * 8;
      float4 q0 = *(const float4*)(qrow + o0), q1 = *(const float4*)(qrow + o0 + 4);
      float4 q2 = *(const float4*)(qrow + o1), q3 = *(const float4*)(qrow + o1 + 4);
      float4 k0 = *(const float4*)(krow + o0), k1 = *(const float4*)(krow + o0 + 4);
      float4 k2 = *(const float4*)(krow + o1), k3 = *(const float4*)(krow + o1 + 4);
      float4 p0 = *(const float4*)(prow + o0), p1 = *(const float4*)(prow + o0 + 4);
      float4 p2 = *(const float4*)(prow + o1), p3 = *(const float4*)(prow + o1 + 4);
      av[0]  = (_Float16)(q0.x - k0.x + p0.x); av[1]  = (_Float16)(q0.y - k0.y + p0.y);
      av[2]  = (_Float16)(q0.z - k0.z + p0.z); av[3]  = (_Float16)(q0.w - k0.w + p0.w);
      av[4]  = (_Float16)(q1.x - k1.x + p1.x); av[5]  = (_Float16)(q1.y - k1.y + p1.y);
      av[6]  = (_Float16)(q1.z - k1.z + p1.z); av[7]  = (_Float16)(q1.w - k1.w + p1.w);
      av[8]  = (_Float16)(q2.x - k2.x + p2.x); av[9]  = (_Float16)(q2.y - k2.y + p2.y);
      av[10] = (_Float16)(q2.z - k2.z + p2.z); av[11] = (_Float16)(q2.w - k2.w + p2.w);
      av[12] = (_Float16)(q3.x - k3.x + p3.x); av[13] = (_Float16)(q3.y - k3.y + p3.y);
      av[14] = (_Float16)(q3.z - k3.z + p3.z); av[15] = (_Float16)(q3.w - k3.w + p3.w);
    } else if (KAL) {
      if (kb + 32 < Kd) __builtin_prefetch(arow + kb + 32, 0, 1);
      const int o0 = kb + hl * 8, o1 = kb + 16 + hl * 8;
      float4 c0 = *(const float4*)(arow + o0), c1 = *(const float4*)(arow + o0 + 4);
      float4 c2 = *(const float4*)(arow + o1), c3 = *(const float4*)(arow + o1 + 4);
      av[0]  = (_Float16)c0.x; av[1]  = (_Float16)c0.y; av[2]  = (_Float16)c0.z; av[3]  = (_Float16)c0.w;
      av[4]  = (_Float16)c1.x; av[5]  = (_Float16)c1.y; av[6]  = (_Float16)c1.z; av[7]  = (_Float16)c1.w;
      av[8]  = (_Float16)c2.x; av[9]  = (_Float16)c2.y; av[10] = (_Float16)c2.z; av[11] = (_Float16)c2.w;
      av[12] = (_Float16)c3.x; av[13] = (_Float16)c3.y; av[14] = (_Float16)c3.z; av[15] = (_Float16)c3.w;
    } else {
#pragma unroll
      for (int j = 0; j < 8; ++j) {
        int ka = kb + hl * 8 + j;
        int kc = kb + 16 + hl * 8 + j;
        av[j]     = (_Float16)((ka < Kd) ? arow[ka] : 0.f);
        av[8 + j] = (_Float16)((kc < Kd) ? arow[kc] : 0.f);
      }
    }
    // ---- B fragments from packed weights (padded => branch free)
    // pair v holds K = kb + (lane>>4)*8 + 2v (v<4), kb + 16 + (lane>>4)*8 + 2(v-4) (v>=4)
    const unsigned* wp = Wp + (size_t)((kb + hl * 8) >> 1) * Nd;
    BFrag bf[NT];
#pragma unroll
    for (int nt = 0; nt < NT; ++nt) {
#pragma unroll
      for (int v = 0; v < 8; ++v)
        bf[nt].u[v] = wp[(size_t)(v < 4 ? v : v + 4) * Nd + col0 + nt * 16];
    }
#pragma unroll
    for (int nt = 0; nt < NT; ++nt)
      acc[nt] = __builtin_amdgcn_wmma_f32_16x16x32_f16(false, av, false, bf[nt].h,
                                                       (short)0, acc[nt], false, false);
  }

  // C/D: lane holds column lane&15; VGPR r holds row (lane>>4)*8 + r.
#pragma unroll
  for (int nt = 0; nt < NT; ++nt) {
    const int c = col0 + nt * 16;
#pragma unroll
    for (int r = 0; r < 8; ++r) {
      int orow = tileM + hl * 8 + r;
      float v0 = acc[nt][r];
      if (bias)    v0 += bias[c];
      if (resid)   v0 += resid[(size_t)orow * Nd + c];
      if (do_relu) v0 = fmaxf(v0, 0.f);
      out[(size_t)orow * Nd + c] = v0;
    }
  }
}

// --------------------------- helper kernels --------------------------------

// x (B,3,N) -> xin (B,N,3)
__global__ void transpose_in_kernel(const float* __restrict__ x, float* __restrict__ xin,
                                    int B, int N)
{
  int t = blockIdx.x * blockDim.x + threadIdx.x;
  if (t >= B * N * 3) return;
  int c = t % 3, n = (t / 3) % N, b = t / (3 * N);
  xin[t] = x[((size_t)b * 3 + c) * N + n];
}

// brute-force kNN (stable ties => matches argsort[:, :, :k])
__global__ void knn_kernel(const float* __restrict__ qxyz, const float* __restrict__ rxyz,
                           int* __restrict__ idx, int B, int Nq, int Nr, int kn)
{
  int t = blockIdx.x * blockDim.x + threadIdx.x;
  if (t >= B * Nq) return;
  int b = t / Nq, n = t % Nq;
  const float* qp = qxyz + ((size_t)b * Nq + n) * 3;
  float qx = qp[0], qy = qp[1], qz = qp[2];
  float bestd[KMAX]; int besti[KMAX];
  for (int i = 0; i < KMAX; ++i) { bestd[i] = 3.4e38f; besti[i] = 0; }
  const float* rp = rxyz + (size_t)b * Nr * 3;
  for (int m = 0; m < Nr; ++m) {
    float dx = qx - rp[3 * m], dy = qy - rp[3 * m + 1], dz = qz - rp[3 * m + 2];
    float d = dx * dx + dy * dy + dz * dz;
    if (d < bestd[kn - 1]) {
      int p = kn - 1;
      while (p > 0 && d < bestd[p - 1]) { bestd[p] = bestd[p - 1]; besti[p] = besti[p - 1]; --p; }
      bestd[p] = d; besti[p] = m;
    }
  }
  int* op = idx + ((size_t)b * Nq + n) * kn;
  for (int i = 0; i < kn; ++i) op[i] = besti[i];
}

// farthest point sampling: one block per batch
__global__ __launch_bounds__(256)
void fps_kernel(const float* __restrict__ xyz, float* __restrict__ dist,
                int* __restrict__ cents, int N, int npoint)
{
  int b = blockIdx.x, tid = threadIdx.x;
  __shared__ float sv[256];
  __shared__ int   si[256];
  __shared__ int   sfar;
  const float* xb = xyz + (size_t)b * N * 3;
  float* db = dist + (size_t)b * N;
  for (int n = tid; n < N; n += 256) db[n] = 1e10f;
  if (tid == 0) sfar = 0;
  __syncthreads();
  for (int it = 0; it < npoint; ++it) {
    int far = sfar;
    if (tid == 0) cents[(size_t)b * npoint + it] = far;
    float cx = xb[3 * far], cy = xb[3 * far + 1], cz = xb[3 * far + 2];
    float bv = -1.f; int bi = 0x7fffffff;
    for (int n = tid; n < N; n += 256) {
      float dx = xb[3 * n] - cx, dy = xb[3 * n + 1] - cy, dz = xb[3 * n + 2] - cz;
      float d = dx * dx + dy * dy + dz * dz;
      float dm = fminf(db[n], d);
      db[n] = dm;
      if (dm > bv) { bv = dm; bi = n; }   // n strictly increasing -> first max kept
    }
    sv[tid] = bv; si[tid] = bi;
    __syncthreads();
    for (int s = 128; s > 0; s >>= 1) {
      if (tid < s) {
        if (sv[tid + s] > sv[tid] || (sv[tid + s] == sv[tid] && si[tid + s] < si[tid])) {
          sv[tid] = sv[tid + s]; si[tid] = si[tid + s];
        }
      }
      __syncthreads();
    }
    if (tid == 0) sfar = si[0];
    __syncthreads();
  }
}

// rel[b,n,j,:] = xyz[b,n,:] - xyz[b,idx[b,n,j],:]
__global__ void rel_kernel(const float* __restrict__ xyz, const int* __restrict__ idx,
                           float* __restrict__ rel, int B, int N, int kn, int total)
{
  int t = blockIdx.x * blockDim.x + threadIdx.x;
  if (t >= total) return;
  int n = (t / kn) % N, b = t / (kn * N);
  int nb = idx[t];
  const float* p  = xyz + ((size_t)b * N + n) * 3;
  const float* pn = xyz + ((size_t)b * N + nb) * 3;
  float* o = rel + (size_t)t * 3;
  o[0] = p[0] - pn[0]; o[1] = p[1] - pn[1]; o[2] = p[2] - pn[2];
}

// softmax over neighbors (axis K) of logits/scale, then sum attn*(v_gather+pos)
__global__ void softmax_res_kernel(const float* __restrict__ logits, const float* __restrict__ vf,
                                   const float* __restrict__ pos, const int* __restrict__ idx,
                                   float* __restrict__ res, int B, int N, int kn)
{
  const int D = 768;
  int t = blockIdx.x * blockDim.x + threadIdx.x;
  if (t >= B * N * D) return;
  int c = t % D, n = (t / D) % N, b = t / (D * N);
  const float inv_scale = 0.03608439182435161f; // 1/sqrt(768)
  size_t base = ((size_t)b * N + n) * kn * D + c;
  float m = -3.4e38f;
  for (int j = 0; j < kn; ++j) m = fmaxf(m, logits[base + (size_t)j * D] * inv_scale);
  float e[KMAX], s = 0.f;
  for (int j = 0; j < kn; ++j) { e[j] = __expf(logits[base + (size_t)j * D] * inv_scale - m); s += e[j]; }
  float inv_s = 1.f / s;
  const int* ip = idx + ((size_t)b * N + n) * kn;
  float r = 0.f;
  for (int j = 0; j < kn; ++j) {
    int nb = ip[j];
    float vv = vf[((size_t)b * N + nb) * D + c] + pos[base + (size_t)j * D];
    r += e[j] * inv_s * vv;
  }
  res[t] = r;
}

// h[b,n,j,:] = concat(xyz[idx]-new_xyz, pts[idx])
__global__ void group_concat_kernel(const float* __restrict__ xyz, const float* __restrict__ nxyz,
                                    const float* __restrict__ pts, const int* __restrict__ idx,
                                    float* __restrict__ h, int B, int N, int Np, int Cin, int total)
{
  int t = blockIdx.x * blockDim.x + threadIdx.x;
  if (t >= total) return;
  int inch = Cin + 3;
  int c = t % inch, j = (t / inch) % KMAX, n = (t / (inch * KMAX)) % Np, b = t / (inch * KMAX * Np);
  int nb = idx[((size_t)b * Np + n) * KMAX + j];
  float v;
  if (c < 3) v = xyz[((size_t)b * N + nb) * 3 + c] - nxyz[((size_t)b * Np + n) * 3 + c];
  else       v = pts[((size_t)b * N + nb) * Cin + (c - 3)];
  h[t] = v;
}

__global__ void gather_xyz_kernel(const float* __restrict__ xyz, const int* __restrict__ cents,
                                  float* __restrict__ nxyz, int B, int N, int Np)
{
  int t = blockIdx.x * blockDim.x + threadIdx.x;
  if (t >= B * Np * 3) return;
  int c = t % 3, n = (t / 3) % Np, b = t / (3 * Np);
  nxyz[t] = xyz[((size_t)b * N + cents[(size_t)b * Np + n]) * 3 + c];
}

__global__ __launch_bounds__(256)
void bn_stats_kernel(const float* __restrict__ h, float* __restrict__ mean,
                     float* __restrict__ var, int Mrows, int C)
{
  int c = blockIdx.x, tid = threadIdx.x;
  __shared__ float ss[256], sq[256];
  float s = 0.f, q = 0.f;
  for (int r = tid; r < Mrows; r += 256) {
    float v = h[(size_t)r * C + c];
    s += v; q += v * v;
  }
  ss[tid] = s; sq[tid] = q;
  __syncthreads();
  for (int st = 128; st > 0; st >>= 1) {
    if (tid < st) { ss[tid] += ss[tid + st]; sq[tid] += sq[tid + st]; }
    __syncthreads();
  }
  if (tid == 0) {
    float m = ss[0] / (float)Mrows;
    mean[c] = m;
    var[c] = sq[0] / (float)Mrows - m * m;
  }
}

__global__ void bn_apply_kernel(float* __restrict__ h, const float* __restrict__ mean,
                                const float* __restrict__ var, const float* __restrict__ g,
                                const float* __restrict__ bb, int total, int C)
{
  int t = blockIdx.x * blockDim.x + threadIdx.x;
  if (t >= total) return;
  int c = t % C;
  float v = (h[t] - mean[c]) * rsqrtf(var[c] + 1e-5f) * g[c] + bb[c];
  h[t] = fmaxf(v, 0.f);
}

// out[r,c] = max_j h[r,j,c]
__global__ void maxk_kernel(const float* __restrict__ h, float* __restrict__ out, int BNp, int C)
{
  int t = blockIdx.x * blockDim.x + threadIdx.x;
  if (t >= BNp * C) return;
  int c = t % C, r = t / C;
  float m = -3.4e38f;
  for (int j = 0; j < KMAX; ++j) m = fmaxf(m, h[((size_t)r * KMAX + j) * C + c]);
  out[t] = m;
}

__global__ void mean_pts_kernel(const float* __restrict__ pts, float* __restrict__ out,
                                int B, int N, int C)
{
  int t = blockIdx.x * blockDim.x + threadIdx.x;
  if (t >= B * C) return;
  int c = t % C, b = t / C;
  float s = 0.f;
  for (int n = 0; n < N; ++n) s += pts[((size_t)b * N + n) * C + c];
  out[t] = s / (float)N;
}

// --------------------------- host orchestration ----------------------------

struct Lin { const float* w; const float* b; };
struct BNp { const float* g; const float* b; };
struct TBlock { Lin delta1, delta2, fc1, fc2, gamma1, gamma2; const float* wq; const float* wk; const float* wv; };
struct TDp { BNp bn1, bn2; Lin conv1, conv2; };

static void gemm(hipStream_t st, unsigned* wpack,
                 const float* A, const float* W, const float* bias,
                 const float* resid, float* out, int M, int Kd, int Nd, int relu)
{
  int Kpad = (Kd + 31) & ~31;
  int ptot = (Kpad >> 1) * Nd;
  pack_w_kernel<<<cdiv_i(ptot, 256), 256, 0, st>>>(W, wpack, Kd, Nd, Kpad);
  bool kal = (Kd % 32 == 0);
  if (Nd % 64 == 0) {
    dim3 g(M / 16, Nd / 64);
    if (kal)
      gemm_wmma_t<0, true, 4><<<g, 32, 0, st>>>(A, wpack, bias, resid, out, M, Kd, Nd, relu,
                                                nullptr, nullptr, nullptr, nullptr, 0, 0);
    else
      gemm_wmma_t<0, false, 4><<<g, 32, 0, st>>>(A, wpack, bias, resid, out, M, Kd, Nd, relu,
                                                 nullptr, nullptr, nullptr, nullptr, 0, 0);
  } else {
    dim3 g(M / 16, Nd / 32);
    if (kal)
      gemm_wmma_t<0, true, 2><<<g, 32, 0, st>>>(A, wpack, bias, resid, out, M, Kd, Nd, relu,
                                                nullptr, nullptr, nullptr, nullptr, 0, 0);
    else
      gemm_wmma_t<0, false, 2><<<g, 32, 0, st>>>(A, wpack, bias, resid, out, M, Kd, Nd, relu,
                                                 nullptr, nullptr, nullptr, nullptr, 0, 0);
  }
}

static void gemm_fused_qkp(hipStream_t st, unsigned* wpack,
                           const float* qv, const float* kv, const float* pv,
                           const int* idx, const float* W, const float* bias, float* out,
                           int M, int Nd, int relu, int Npts, int kn)
{
  int ptot = (768 >> 1) * Nd;
  pack_w_kernel<<<cdiv_i(ptot, 256), 256, 0, st>>>(W, wpack, 768, Nd, 768);
  dim3 g(M / 16, Nd / 64);   // Nd = 768 here
  gemm_wmma_t<1, true, 4><<<g, 32, 0, st>>>(nullptr, wpack, bias, nullptr, out, M, 768, Nd, relu,
                                            qv, kv, pv, idx, Npts, kn);
}

static void run_transformer(hipStream_t st, const TBlock& tp, unsigned* wpack,
                            const float* xyz, const float* pts_in, float* pts_out,
                            int B, int N, int Dp,
                            int* idx, float* xb, float* qb, float* kbuf, float* vb,
                            float* relb, float* big1, float* big2, float* big3, float* resb)
{
  const int kn  = (N < KMAX) ? N : KMAX;       // argsort[..., :20] on axis of size N
  const int BN  = B * N;
  const int BNK = BN * kn;
  knn_kernel<<<cdiv_i(BN, 128), 128, 0, st>>>(xyz, xyz, idx, B, N, N, kn);
  gemm(st, wpack, pts_in, tp.fc1.w, tp.fc1.b, nullptr, xb, BN, Dp, 768, 0);
  gemm(st, wpack, xb, tp.wq, nullptr, nullptr, qb,   BN, 768, 768, 0);
  gemm(st, wpack, xb, tp.wk, nullptr, nullptr, kbuf, BN, 768, 768, 0);
  gemm(st, wpack, xb, tp.wv, nullptr, nullptr, vb,   BN, 768, 768, 0);
  rel_kernel<<<cdiv_i(BNK, 256), 256, 0, st>>>(xyz, idx, relb, B, N, kn, BNK);
  gemm(st, wpack, relb, tp.delta1.w, tp.delta1.b, nullptr, big1, BNK, 3,   768, 1); // relu(delta1)
  gemm(st, wpack, big1, tp.delta2.w, tp.delta2.b, nullptr, big2, BNK, 768, 768, 0); // pos
  gemm_fused_qkp(st, wpack, qb, kbuf, big2, idx, tp.gamma1.w, tp.gamma1.b, big1, BNK, 768, 1, N, kn);
  gemm(st, wpack, big1, tp.gamma2.w, tp.gamma2.b, nullptr, big3, BNK, 768, 768, 0); // logits
  softmax_res_kernel<<<cdiv_i(BN * 768, 256), 256, 0, st>>>(big3, vb, big2, idx, resb, B, N, kn);
  gemm(st, wpack, resb, tp.fc2.w, tp.fc2.b, pts_in, pts_out, BN, 768, Dp, 0);       // + residual
}

static void run_td(hipStream_t st, const TDp& p, unsigned* wpack,
                   const float* xyz, const float* pts_in,
                   int B, int N, int Cin, int np, int c,
                   float* new_xyz, float* pts_out,
                   int* cents, float* fdist, int* idx,
                   float* hbuf, float* h1, float* h2, float* meanb, float* varb)
{
  fps_kernel<<<B, 256, 0, st>>>(xyz, fdist, cents, N, np);
  gather_xyz_kernel<<<cdiv_i(B * np * 3, 256), 256, 0, st>>>(xyz, cents, new_xyz, B, N, np);
  knn_kernel<<<cdiv_i(B * np, 128), 128, 0, st>>>(new_xyz, xyz, idx, B, np, N, KMAX);
  const int inch = Cin + 3;
  const int tot  = B * np * KMAX * inch;
  group_concat_kernel<<<cdiv_i(tot, 256), 256, 0, st>>>(xyz, new_xyz, pts_in, idx, hbuf,
                                                        B, N, np, Cin, tot);
  const int Mr = B * np * KMAX;
  gemm(st, wpack, hbuf, p.conv1.w, p.conv1.b, nullptr, h1, Mr, inch, c, 0);
  bn_stats_kernel<<<c, 256, 0, st>>>(h1, meanb, varb, Mr, c);
  bn_apply_kernel<<<cdiv_i(Mr * c, 256), 256, 0, st>>>(h1, meanb, varb, p.bn1.g, p.bn1.b, Mr * c, c);
  gemm(st, wpack, h1, p.conv2.w, p.conv2.b, nullptr, h2, Mr, c, c, 0);
  bn_stats_kernel<<<c, 256, 0, st>>>(h2, meanb, varb, Mr, c);
  bn_apply_kernel<<<cdiv_i(Mr * c, 256), 256, 0, st>>>(h2, meanb, varb, p.bn2.g, p.bn2.b, Mr * c, c);
  maxk_kernel<<<cdiv_i(B * np * c, 256), 256, 0, st>>>(h2, pts_out, B * np, c);
}

extern "C" void kernel_launch(void* const* d_in, const int* in_sizes, int n_in,
                              void* d_out, int out_size, void* d_ws, size_t ws_size,
                              hipStream_t stream)
{
  (void)in_sizes; (void)n_in; (void)out_size; (void)ws_size;
  const int B = 2, N0 = 2048, D = 768;
  const int NPTS[4] = {512, 128, 32, 8};
  const int CH[4]   = {64, 128, 256, 512};

  // ---- unpack params: d_in[0]=x, then JAX pytree leaf order (keys sorted).
  const float* xraw = (const float*)d_in[0];
  int pi = 1;
  auto nxt    = [&]() { return (const float*)d_in[pi++]; };
  auto getLin = [&]() { Lin l; l.b = nxt(); l.w = nxt(); return l; };            // 'b' < 'w'
  auto getBN  = [&]() { BNp b; b.b = nxt(); b.g = nxt(); return b; };            // 'b' < 'g'
  auto getTB  = [&]() { TBlock t;
                        t.delta1 = getLin(); t.delta2 = getLin();
                        t.fc1 = getLin(); t.fc2 = getLin();
                        t.gamma1 = getLin(); t.gamma2 = getLin();
                        t.wq = nxt(); t.wk = nxt(); t.wv = nxt(); return t; };
  auto getTD  = [&]() { TDp d; d.bn1 = getBN(); d.bn2 = getBN();
                        d.conv1 = getLin(); d.conv2 = getLin(); return d; };
  TDp    tdp[4];
  TBlock tfp[4];
  for (int i = 0; i < 4; ++i) { tdp[i] = getTD(); tfp[i] = getTB(); }            // 'blocks'
  Lin fc1a = getLin();                                                            // 'fc1a'
  Lin fc1b = getLin();                                                            // 'fc1b'
  TBlock t0 = getTB();                                                            // 't0'

  // ---- workspace layout (floats)
  float* f = (float*)d_ws;
  size_t off = 0;
  auto take = [&](size_t n) { float* p = f + off; off += n; return p; };
  float* xin   = take((size_t)B * N0 * 3);
  float* xyzA  = take((size_t)B * N0 * 3);
  float* xyzB  = take((size_t)B * N0 * 3);
  float* P0    = take((size_t)B * N0 * 32);   // pts ping buffer (max elems)
  float* P1    = take((size_t)B * N0 * 32);   // pts pong buffer
  int*   idxb  = (int*)take((size_t)B * N0 * KMAX);
  int*   cents = (int*)take((size_t)B * 512);
  float* fdist = take((size_t)B * N0);
  float* meanb = take(512);
  float* varb  = take(512);
  unsigned* wpack = (unsigned*)take((size_t)(768 / 2) * 768);  // packed f16 weights
  float* xb    = take((size_t)B * N0 * D);
  float* qb    = take((size_t)B * N0 * D);
  float* kbuf  = take((size_t)B * N0 * D);
  float* vb    = take((size_t)B * N0 * D);
  float* resb  = take((size_t)B * N0 * D);
  float* relb  = take((size_t)B * N0 * KMAX * 3);
  float* big1  = take((size_t)B * N0 * KMAX * D);
  float* big2  = take((size_t)B * N0 * KMAX * D);
  float* big3  = take((size_t)B * N0 * KMAX * D);

  // ---- input MLP: xin = transpose(x); feats = fc1b(relu(fc1a(xin)))
  transpose_in_kernel<<<cdiv_i(B * N0 * 3, 256), 256, 0, stream>>>(xraw, xin, B, N0);
  gemm(stream, wpack, xin, fc1a.w, fc1a.b, nullptr, P1, B * N0, 3, 32, 1);
  gemm(stream, wpack, P1, fc1b.w, fc1b.b, nullptr, P0, B * N0, 32, 32, 0);

  // ---- t0 transformer at full resolution
  run_transformer(stream, t0, wpack, xin, P0, P1, B, N0, 32,
                  idxb, xb, qb, kbuf, vb, relb, big1, big2, big3, resb);

  // ---- 4 stages of (transition-down, transformer)
  const float* cur_xyz = xin;
  int curN = N0, curC = 32;
  for (int s = 0; s < 4; ++s) {
    float* nxyz = (s & 1) ? xyzB : xyzA;
    run_td(stream, tdp[s], wpack, cur_xyz, P1, B, curN, curC, NPTS[s], CH[s],
           nxyz, P0, cents, fdist, idxb, big1, big2, big3, meanb, varb);
    run_transformer(stream, tfp[s], wpack, nxyz, P0, P1, B, NPTS[s], CH[s],
                    idxb, xb, qb, kbuf, vb, relb, big1, big2, big3, resb);
    cur_xyz = nxyz;
    curN = NPTS[s];
    curC = CH[s];
  }

  // ---- global mean pool -> (B, 512)
  mean_pts_kernel<<<cdiv_i(B * 512, 256), 256, 0, stream>>>(P1, (float*)d_out, B, 8, 512);
}
